// Qwen2VLVisionBlock_72541997629649
// MI455X (gfx1250) — compile-verified, matrix operand order checked by reading, not run on hardware
//
#include <hip/hip_runtime.h>
#include <hip/hip_bf16.h>
#include <math.h>

// ---------------- problem constants ----------------
#define N_TOK 2048
#define DIM   1280
#define HEADS 16
#define HD    80
#define HDP   96          // HD padded to a multiple of 32 for WMMA K
#define HID   5120
#define EPSLN 1e-6f

typedef __attribute__((ext_vector_type(16))) _Float16 v16h;
typedef __attribute__((ext_vector_type(8)))  _Float16 v8h;
typedef __attribute__((ext_vector_type(8)))  float    v8f;

// s_wait_asynccnt 0 : wait for this wave's async->LDS copies to land
#if defined(__has_builtin)
#  if __has_builtin(__builtin_amdgcn_s_wait_asynccnt)
#    define ASYNC_WAIT() __builtin_amdgcn_s_wait_asynccnt(0)
#  endif
#endif
#ifndef ASYNC_WAIT
#  define ASYNC_WAIT() asm volatile("s_wait_asynccnt 0x0" ::: "memory")
#endif

// Raw split workgroup barrier: avoids __syncthreads()'s implied full vmem
// drain. LDS ordering is provided by data deps (ds_loads consumed by WMMAs)
// and ASYNC_WAIT before the signal.
#if defined(__has_builtin)
#  if __has_builtin(__builtin_amdgcn_s_barrier_signal) && \
      __has_builtin(__builtin_amdgcn_s_barrier_wait)
#    define WG_BARRIER() do { __builtin_amdgcn_s_barrier_signal(-1); \
                              __builtin_amdgcn_s_barrier_wait((short)-1); } while (0)
#  endif
#endif
#ifndef WG_BARRIER
#  define WG_BARRIER() __syncthreads()
#endif

#define SHUF16(lo, hi) __builtin_shufflevector(lo, hi, \
    0,1,2,3,4,5,6,7,8,9,10,11,12,13,14,15)

// ---------------------------------------------------------------------------
// Generic WMMA GEMM: C(MxN) = A(MxK) * Bt(NxK)^T, f16 in, f32 accum.
// Block = 256 threads = 8 waves; each wave computes (MT*16) rows x (NT*16) cols.
// BOTH A (MT*128 x 32) and B (NT*16 x 32) K-step tiles are staged to LDS with
// global_load_async_to_lds_b128, double buffered: ASYNCcnt decouples the copy
// of step k+32 from the WMMAs consuming step k, and the loop carries no vector
// state except the accumulators (no register rotation / hazard NOPs).
// Fragment layout (ISA 7.12.2, 16-bit A 16x32; B == A-layout of B^T):
//   lanes 0-15  : row=lane,    elems 0..7 = K+0..7,  8..15 = K+16..23
//   lanes 16-31 : row=lane-16, elems 0..7 = K+8..15, 8..15 = K+24..31
// EPI: 0 = f32 +bias            (QKV raw)
//      1 = f32 +bias +residual  (proj / fc2)
//      2 = f16 gelu(acc+bias)   (fc1)
//      3 = f32 acc*p0 + segmask (attention scores)
//      4 = f16 plain            (attn*V scatter; outH pre-offset by head col)
// ---------------------------------------------------------------------------
template<int EPI, int NT, int MT>
__global__ void __launch_bounds__(256)
gemm_wmma(const _Float16* __restrict__ A, int lda,
          const _Float16* __restrict__ Bt, int ldb,
          const float* __restrict__ bias,
          const float* __restrict__ res,
          const int* __restrict__ cu, int ncu,
          float* __restrict__ outF, _Float16* __restrict__ outH, int ldo,
          int M, int N, int K, float p0)
{
  constexpr int BROWS = NT * 16;    // B-tile rows (output columns)
  constexpr int AROWS = MT * 128;   // A-tile rows (8 waves * MT*16)
  constexpr int LSTR  = 40;         // f16 elems per LDS row (80B, 16B aligned)
  __shared__ _Float16 smB[2][BROWS * LSTR];
  __shared__ _Float16 smA[2][AROWS * LSTR];

  const int lane = threadIdx.x & 31;
  const int wave = threadIdx.x >> 5;
  const int mB   = blockIdx.y * AROWS;       // block A-row base
  const int m0   = mB + wave * (MT * 16);
  const int n0   = blockIdx.x * BROWS;
  const int r    = lane & 15;
  const int koff = (lane & 16) ? 8 : 0;

  // async-copy A+B tiles for K-step k0 into buffers [buf]
  auto stage = [&](int buf, int k0) {
#pragma unroll
    for (int c0 = 0; c0 < BROWS * 4; c0 += 256) {
      const int c = c0 + threadIdx.x;
      if (c < BROWS * 4) {
        const int row = c >> 2, ch = c & 3;            // 4x16B chunks per row
        const _Float16* src = Bt + (size_t)(n0 + row) * ldb + k0 + ch * 8;
        unsigned dst = (unsigned)(size_t)&smB[buf][row * LSTR + ch * 8];
        unsigned long long ga = (unsigned long long)(size_t)src;
        asm volatile("global_load_async_to_lds_b128 %0, %1, off"
                     :: "v"(dst), "v"(ga) : "memory");
      }
    }
#pragma unroll
    for (int c0 = 0; c0 < AROWS * 4; c0 += 256) {      // AROWS*4 % 256 == 0
      const int c = c0 + threadIdx.x;
      const int row = c >> 2, ch = c & 3;
      const _Float16* src = A + (size_t)(mB + row) * lda + k0 + ch * 8;
      unsigned dst = (unsigned)(size_t)&smA[buf][row * LSTR + ch * 8];
      unsigned long long ga = (unsigned long long)(size_t)src;
      asm volatile("global_load_async_to_lds_b128 %0, %1, off"
                   :: "v"(dst), "v"(ga) : "memory");
    }
  };

  v8f acc[MT][NT] = {};
  stage(0, 0);

  int buf = 0;
  for (int k0 = 0; k0 < K; k0 += 32, buf ^= 1) {
    ASYNC_WAIT();        // own chunks of tiles[buf] landed
    WG_BARRIER();        // everyone's chunks landed; prev readers done
    if (k0 + 32 < K) stage(buf ^ 1, k0 + 32);   // copy next tiles now

    // batch all LDS fragment loads before the WMMA chain
    v8h al[MT], ah[MT], bl[NT], bh[NT];
#pragma unroll
    for (int mm = 0; mm < MT; ++mm) {
      const _Float16* ap = &smA[buf][(wave * MT * 16 + mm * 16 + r) * LSTR + koff];
      al[mm] = *(const v8h*)(ap);
      ah[mm] = *(const v8h*)(ap + 16);
    }
#pragma unroll
    for (int t = 0; t < NT; ++t) {
      const _Float16* bp = &smB[buf][(t * 16 + r) * LSTR + koff];
      bl[t] = *(const v8h*)(bp);
      bh[t] = *(const v8h*)(bp + 16);
    }
    v16h af[MT];
#pragma unroll
    for (int mm = 0; mm < MT; ++mm) af[mm] = SHUF16(al[mm], ah[mm]);
#pragma unroll
    for (int t = 0; t < NT; ++t) {
      v16h b = SHUF16(bl[t], bh[t]);
#pragma unroll
      for (int mm = 0; mm < MT; ++mm)
        acc[mm][t] = __builtin_amdgcn_wmma_f32_16x16x32_f16(
            false, af[mm], false, b, (short)0, acc[mm][t], false, false);
    }
  }

  const int rl = (lane >> 4) << 3;   // +8 row offset for upper half-wave
  const int cl = lane & 15;

  int seg[8];
  if constexpr (EPI == 3) {
    for (int j = 0; j < 8; ++j) seg[j] = (j + 1 < ncu) ? cu[j + 1] : 0x7fffffff;
  }
  auto segof = [&](int i) {
    int s = 0;
    for (int j = 0; j + 1 < ncu && j < 8; ++j) s += (seg[j] <= i) ? 1 : 0;
    return s;
  };

#pragma unroll
  for (int mm = 0; mm < MT; ++mm) {
#pragma unroll
    for (int t = 0; t < NT; ++t) {
#pragma unroll
      for (int j = 0; j < 8; ++j) {
        const int m = m0 + mm * 16 + j + rl;
        const int n = n0 + t * 16 + cl;
        const float v = acc[mm][t][j];
        if constexpr (EPI == 0) {
          outF[(size_t)m * ldo + n] = v + bias[n];
        } else if constexpr (EPI == 1) {
          outF[(size_t)m * ldo + n] = v + bias[n] + res[(size_t)m * ldo + n];
        } else if constexpr (EPI == 2) {
          float x = v + bias[n];
          outH[(size_t)m * ldo + n] = (_Float16)(x / (1.f + __expf(-1.702f * x)));
        } else if constexpr (EPI == 3) {
          outF[(size_t)m * ldo + n] =
              v * p0 + ((segof(m) != segof(n)) ? 1.0f : 0.0f);
        } else {
          outH[(size_t)m * ldo + n] = (_Float16)v;
        }
      }
    }
  }
}

// ---------------------------------------------------------------------------
// fp32 (KxN) -> f16 transposed (NxK), LDS tiled
// ---------------------------------------------------------------------------
__global__ void __launch_bounds__(256)
transpose_cvt(const float* __restrict__ w, _Float16* __restrict__ wt, int K, int N)
{
  __shared__ _Float16 t[16][17];
  int k = blockIdx.y * 16 + threadIdx.y;
  int n = blockIdx.x * 16 + threadIdx.x;
  t[threadIdx.y][threadIdx.x] = (_Float16)w[(size_t)k * N + n];
  __syncthreads();
  int on = blockIdx.x * 16 + threadIdx.y;
  int ok = blockIdx.y * 16 + threadIdx.x;
  wt[(size_t)on * K + ok] = t[threadIdx.x][threadIdx.y];
}

// ---------------------------------------------------------------------------
// LayerNorm row -> f16
// ---------------------------------------------------------------------------
__global__ void __launch_bounds__(256)
ln_kernel(const float* __restrict__ x, const float* __restrict__ g,
          const float* __restrict__ b, _Float16* __restrict__ o, int D)
{
  __shared__ float red[256];
  const int row = blockIdx.x, tid = threadIdx.x;
  const float* xr = x + (size_t)row * D;

  float s = 0.f;
  for (int i = tid; i < D; i += 256) s += xr[i];
  red[tid] = s; __syncthreads();
  for (int st = 128; st > 0; st >>= 1) {
    if (tid < st) red[tid] += red[tid + st];
    __syncthreads();
  }
  const float mu = red[0] / (float)D;
  __syncthreads();

  float v = 0.f;
  for (int i = tid; i < D; i += 256) { float d = xr[i] - mu; v += d * d; }
  red[tid] = v; __syncthreads();
  for (int st = 128; st > 0; st >>= 1) {
    if (tid < st) red[tid] += red[tid + st];
    __syncthreads();
  }
  const float rs = rsqrtf(red[0] / (float)D + EPSLN);

  for (int i = tid; i < D; i += 256)
    o[(size_t)row * D + i] = (_Float16)((xr[i] - mu) * rs * g[i] + b[i]);
}

// ---------------------------------------------------------------------------
// Row softmax: f32 scores -> f16 probabilities
// ---------------------------------------------------------------------------
__global__ void __launch_bounds__(256)
softmax_kernel(const float* __restrict__ S, _Float16* __restrict__ P, int Nc)
{
  __shared__ float red[256];
  const int row = blockIdx.x, tid = threadIdx.x;
  const float* sr = S + (size_t)row * Nc;

  float mx = -3.4e38f;
  for (int i = tid; i < Nc; i += 256) mx = fmaxf(mx, sr[i]);
  red[tid] = mx; __syncthreads();
  for (int st = 128; st > 0; st >>= 1) {
    if (tid < st) red[tid] = fmaxf(red[tid], red[tid + st]);
    __syncthreads();
  }
  mx = red[0]; __syncthreads();

  float sum = 0.f;
  for (int i = tid; i < Nc; i += 256) sum += __expf(sr[i] - mx);
  red[tid] = sum; __syncthreads();
  for (int st = 128; st > 0; st >>= 1) {
    if (tid < st) red[tid] += red[tid + st];
    __syncthreads();
  }
  const float inv = 1.f / red[0];

  for (int i = tid; i < Nc; i += 256)
    P[(size_t)row * Nc + i] = (_Float16)(__expf(sr[i] - mx) * inv);
}

// ---------------------------------------------------------------------------
// RoPE + repack: qkv raw f32 [n][3*DIM] -> qp/kp f16 [h][n][HDP] (zero pad),
// v -> vT f16 [h][HD][n]
// ---------------------------------------------------------------------------
__global__ void __launch_bounds__(96)
rope_pack(const float* __restrict__ qkv, const float* __restrict__ rpe,
          _Float16* __restrict__ qp, _Float16* __restrict__ kp,
          _Float16* __restrict__ vT)
{
  const int h = blockIdx.x, n = blockIdx.y, d = threadIdx.x;  // d in [0,96)
  const size_t base = (size_t)n * (3 * DIM) + (size_t)h * HD;

  _Float16 qo = (_Float16)0.f, ko = (_Float16)0.f;
  if (d < HD) {
    const int dm = (d < HD / 2) ? d : d - HD / 2;
    const float ang = rpe[(size_t)n * (HD / 2) + dm];
    const float cs = __cosf(ang), sn = __sinf(ang);
    const float q = qkv[base + d];
    const float k = qkv[base + DIM + d];
    const float qr = (d < HD / 2) ? -qkv[base + d + HD / 2] : qkv[base + d - HD / 2];
    const float kr = (d < HD / 2) ? -qkv[base + DIM + d + HD / 2] : qkv[base + DIM + d - HD / 2];
    qo = (_Float16)(q * cs + qr * sn);
    ko = (_Float16)(k * cs + kr * sn);
    vT[((size_t)h * HD + d) * N_TOK + n] = (_Float16)qkv[base + 2 * DIM + d];
  }
  const size_t po = ((size_t)h * N_TOK + n) * HDP + d;
  qp[po] = qo;
  kp[po] = ko;
}

// ---------------- workspace layout (bytes, all 256-aligned) ----------------
#define O_WQKV  ((size_t)0)                       // 3840*1280*2  = 9,830,400
#define O_WPROJ ((size_t)9830400)                 // 1280*1280*2  = 3,276,800
#define O_WFC1  ((size_t)13107200)                // 5120*1280*2  = 13,107,200
#define O_WFC2  ((size_t)26214400)                // 1280*5120*2  = 13,107,200
#define O_XN    ((size_t)39321600)                // 2048*1280*2  (ln1 out; reused for ln2 out)
#define O_ATTO  ((size_t)44564480)                // 2048*1280*2  attn output f16
#define O_H     ((size_t)49807360)                // 2048*1280*4  residual h f32
#define O_BUFA  ((size_t)60293120)                // 2048*3840*4  qkv raw; later S (16MB) + P (8MB)
#define O_S     (O_BUFA)
#define O_P     (O_BUFA + (size_t)16777216)
#define O_BUFB  ((size_t)91750400)                // packs (17.8MB) then fc1 act f16 (21MB)
#define O_QP    (O_BUFB)
#define O_KP    (O_BUFB + (size_t)6291456)
#define O_VT    (O_BUFB + (size_t)12582912)
#define O_ACT   (O_BUFB)

extern "C" void kernel_launch(void* const* d_in, const int* in_sizes, int n_in,
                              void* d_out, int out_size, void* d_ws, size_t ws_size,
                              hipStream_t stream) {
  const float* hidden = (const float*)d_in[0];
  const float* rpe    = (const float*)d_in[1];
  const int*   cu     = (const int*)d_in[2];
  const float* qkv_w  = (const float*)d_in[3];
  const float* qkv_b  = (const float*)d_in[4];
  const float* proj_w = (const float*)d_in[5];
  const float* proj_b = (const float*)d_in[6];
  const float* fc1_w  = (const float*)d_in[7];
  const float* fc1_b  = (const float*)d_in[8];
  const float* fc2_w  = (const float*)d_in[9];
  const float* fc2_b  = (const float*)d_in[10];
  const float* ln1_s  = (const float*)d_in[11];
  const float* ln1_b  = (const float*)d_in[12];
  const float* ln2_s  = (const float*)d_in[13];
  const float* ln2_b  = (const float*)d_in[14];
  const int ncu = in_sizes[2];

  unsigned char* ws = (unsigned char*)d_ws;
  _Float16* wqkv  = (_Float16*)(ws + O_WQKV);
  _Float16* wproj = (_Float16*)(ws + O_WPROJ);
  _Float16* wfc1  = (_Float16*)(ws + O_WFC1);
  _Float16* wfc2  = (_Float16*)(ws + O_WFC2);
  _Float16* xn    = (_Float16*)(ws + O_XN);
  _Float16* atto  = (_Float16*)(ws + O_ATTO);
  float*    hbuf  = (float*)   (ws + O_H);
  float*    qkvr  = (float*)   (ws + O_BUFA);
  float*    Sbuf  = (float*)   (ws + O_S);
  _Float16* Pbuf  = (_Float16*)(ws + O_P);
  _Float16* qp    = (_Float16*)(ws + O_QP);
  _Float16* kp    = (_Float16*)(ws + O_KP);
  _Float16* vT    = (_Float16*)(ws + O_VT);
  _Float16* act   = (_Float16*)(ws + O_ACT);
  float*    outp  = (float*)d_out;

  const dim3 tb(16, 16);
  // ---- stage weights: f32 (KxN) -> f16 transposed (NxK) ----
  transpose_cvt<<<dim3(3 * DIM / 16, DIM / 16), tb, 0, stream>>>(qkv_w, wqkv, DIM, 3 * DIM);
  transpose_cvt<<<dim3(DIM / 16,     DIM / 16), tb, 0, stream>>>(proj_w, wproj, DIM, DIM);
  transpose_cvt<<<dim3(HID / 16,     DIM / 16), tb, 0, stream>>>(fc1_w, wfc1, DIM, HID);
  transpose_cvt<<<dim3(DIM / 16,     HID / 16), tb, 0, stream>>>(fc2_w, wfc2, HID, DIM);

  // ---- LN1 ----
  ln_kernel<<<N_TOK, 256, 0, stream>>>(hidden, ln1_s, ln1_b, xn, DIM);

  // ---- QKV GEMM: (2048x1280)x(1280x3840) + bias -> f32 raw ----
  gemm_wmma<0, 4, 2><<<dim3(3 * DIM / 64, N_TOK / 256), 256, 0, stream>>>(
      xn, DIM, wqkv, DIM, qkv_b, nullptr, nullptr, 0,
      qkvr, nullptr, 3 * DIM, N_TOK, 3 * DIM, DIM, 0.f);

  // ---- RoPE + per-head repack ----
  rope_pack<<<dim3(HEADS, N_TOK), 96, 0, stream>>>(qkvr, rpe, qp, kp, vT);

  // ---- attention, per head (S/P buffers reused serially) ----
  const float scl = 0.11180339887498949f;  // 1/sqrt(80)
  for (int h = 0; h < HEADS; ++h) {
    const _Float16* qh = qp + (size_t)h * N_TOK * HDP;
    const _Float16* kh = kp + (size_t)h * N_TOK * HDP;
    const _Float16* vh = vT + (size_t)h * HD * N_TOK;
    // scores: Q (2048x96) x K^T -> f32, *scale, +segment mask
    gemm_wmma<3, 4, 2><<<dim3(N_TOK / 64, N_TOK / 256), 256, 0, stream>>>(
        qh, HDP, kh, HDP, nullptr, nullptr, cu, ncu,
        Sbuf, nullptr, N_TOK, N_TOK, N_TOK, HDP, scl);
    // row softmax -> f16 P
    softmax_kernel<<<N_TOK, 256, 0, stream>>>(Sbuf, Pbuf, N_TOK);
    // out = P (2048x2048) x V (2048x80) -> scatter f16 into [n][h*80+d]
    gemm_wmma<4, 5, 2><<<dim3(1, N_TOK / 256), 256, 0, stream>>>(
        Pbuf, N_TOK, vh, N_TOK, nullptr, nullptr, nullptr, 0,
        nullptr, atto + (size_t)h * HD, DIM, N_TOK, HD, N_TOK, 0.f);
  }

  // ---- proj + residual: h = hidden + atto @ proj_w + b ----
  gemm_wmma<1, 4, 2><<<dim3(DIM / 64, N_TOK / 256), 256, 0, stream>>>(
      atto, DIM, wproj, DIM, proj_b, hidden, nullptr, 0,
      hbuf, nullptr, DIM, N_TOK, DIM, DIM, 0.f);

  // ---- LN2 (reuse xn) ----
  ln_kernel<<<N_TOK, 256, 0, stream>>>(hbuf, ln2_s, ln2_b, xn, DIM);

  // ---- FC1 + fast-gelu -> f16 act ----
  gemm_wmma<2, 4, 2><<<dim3(HID / 64, N_TOK / 256), 256, 0, stream>>>(
      xn, DIM, wfc1, DIM, fc1_b, nullptr, nullptr, 0,
      nullptr, act, HID, N_TOK, HID, DIM, 0.f);

  // ---- FC2 + bias + residual h -> d_out f32 ----
  gemm_wmma<1, 4, 2><<<dim3(DIM / 64, N_TOK / 256), 256, 0, stream>>>(
      act, HID, wfc2, HID, fc2_b, hbuf, nullptr, 0,
      outp, nullptr, DIM, N_TOK, DIM, HID, 0.f);
}
